// ConvAttnPool_45870250721434
// MI455X (gfx1250) — compile-verified, matrix operand order checked by reading, not run on hardware
//
#include <hip/hip_runtime.h>
#include <math.h>

typedef _Float16 half_t;
typedef __attribute__((ext_vector_type(16))) _Float16 v16h;
typedef __attribute__((ext_vector_type(8)))  float    v8f;

#define BB 8
#define LL 2048
#define EE 100
#define FF 256
#define KK 9
#define YY 8921
#define YPAD 8928            // 558 * 16
#define YTILES 558
#define KFLAT 900            // E*K
#define KPAD 928             // 29 * 32
#define NYH (BB * YY)        // 71368 yhat elements
#define ALPHA_OFF (NYH + 1)  // alpha starts after yhat + loss

union V16U { v16h v; uint32_t u[8]; };

static __device__ __forceinline__ v8f wmma_f16(v16h a, v16h b, v8f c) {
    return __builtin_amdgcn_wmma_f32_16x16x32_f16(false, a, false, b, (short)0, c, false, false);
}

// ---------------------------------------------------------------------------
// Kprep: U_w (fp32, zero pad to YPAD) -> f16 ; conv_w -> wBT[f][k*E+e] f16 pad
// ---------------------------------------------------------------------------
__global__ void cap_prep(const float* __restrict__ U_w,
                         const float* __restrict__ conv_w,
                         half_t* __restrict__ U_half,
                         half_t* __restrict__ wBT) {
    const int NA = YPAD * FF;
    const int NB = FF * KPAD;
    int i = blockIdx.x * blockDim.x + threadIdx.x;
    if (i < NA) {
        int y = i / FF, f = i % FF;
        U_half[i] = (y < YY) ? (half_t)U_w[y * FF + f] : (half_t)0.0f;
    } else if (i < NA + NB) {
        int j  = i - NA;
        int f  = j / KPAD;
        int kf = j % KPAD;
        half_t v = (half_t)0.0f;
        if (kf < KFLAT) {
            int k = kf / EE, e = kf % EE;
            v = (half_t)conv_w[(f * EE + e) * KK + k];
        }
        wBT[f * KPAD + kf] = v;
    }
}

// ---------------------------------------------------------------------------
// K1: embedding + conv1d(K=9,same) + tanh, as im2col WMMA GEMM.
// block = 256 thr (8 waves), one (b, 32-row L tile) per block.
// ---------------------------------------------------------------------------
__global__ void cap_conv(const int* __restrict__ x,
                         const float* __restrict__ embed_W,
                         const float* __restrict__ conv_b,
                         const half_t* __restrict__ wBT,
                         half_t* __restrict__ h_half,
                         half_t* __restrict__ hT_half) {
    __shared__ half_t emb_s[4064];   // 40 rows x 100, +tail slack (B pad rows are 0)

    const int bid = blockIdx.x;
    const int b   = bid >> 6;          // L/32 = 64 tiles
    const int lt  = bid & 63;
    const int l0  = lt * 32;
    const int tid  = threadIdx.x;
    const int lane = tid & 31;
    const int wv   = tid >> 5;

    // stage 40 embedding rows (l0-4 .. l0+35) as f16
    for (int i = tid; i < 40 * EE; i += 256) {
        int r = i / EE, e = i % EE;
        int gl = l0 + r - 4;
        float v = 0.0f;
        if (gl >= 0 && gl < LL) {
            int idx = x[b * LL + gl];
            v = embed_W[(size_t)idx * EE + e];
        }
        emb_s[i] = (half_t)v;
    }
    __syncthreads();

    const int m0 = (wv & 1) * 16;        // L sub-tile
    const int n0 = (wv >> 1) * 64;       // F range, 4 tiles of 16
    const int lrow  = lane & 15;
    const int khalf = (lane >> 4) * 16;

    v8f c0 = {}, c1 = {}, c2 = {}, c3 = {};
    for (int kk = 0; kk < 29; ++kk) {
        const int k = kk * 32;
        V16U a;
        const uint32_t* ap = (const uint32_t*)&emb_s[(m0 + lrow) * EE + k + khalf];
#pragma unroll
        for (int i = 0; i < 8; ++i) a.u[i] = ap[i];
#pragma unroll
        for (int tt = 0; tt < 4; ++tt) {
            const int fc = n0 + tt * 16 + lrow;
            v16h bfrag = *(const v16h*)&wBT[fc * KPAD + k + khalf];
            v8f cc = (tt == 0) ? c0 : (tt == 1) ? c1 : (tt == 2) ? c2 : c3;
            cc = wmma_f16(a.v, bfrag, cc);
            if (tt == 0) c0 = cc; else if (tt == 1) c1 = cc;
            else if (tt == 2) c2 = cc; else c3 = cc;
        }
    }

    // epilogue: bias + tanh, write h and hT (f16)
#pragma unroll
    for (int tt = 0; tt < 4; ++tt) {
        v8f cc = (tt == 0) ? c0 : (tt == 1) ? c1 : (tt == 2) ? c2 : c3;
        const int f = n0 + tt * 16 + lrow;
        const float bias = conv_b[f];
#pragma unroll
        for (int j = 0; j < 8; ++j) {
            const int l = l0 + m0 + j + ((lane >> 4) << 3);
            half_t hv = (half_t)tanhf(cc[j] + bias);
            h_half [((size_t)b * LL + l) * FF + f] = hv;
            hT_half[((size_t)b * FF + f) * LL + l] = hv;
        }
    }
}

// ---------------------------------------------------------------------------
// K2: scores[b,y,l] = sum_f U[y,f] * h[b,l,f]  (M=YPAD, N=LL, K=FF)
// block 256 thr (8 waves); block tile: 16(M) x 512(N); wave: 16 x 64.
// ---------------------------------------------------------------------------
__global__ void cap_scores(const half_t* __restrict__ U_half,
                           const half_t* __restrict__ h_half,
                           float* __restrict__ alpha) {
    const int bid = blockIdx.x;
    const int lc  = bid & 3;                 // 4 chunks of 512
    const int yt  = (bid >> 2) % YTILES;
    const int b   = bid / (4 * YTILES);
    const int tid  = threadIdx.x;
    const int lane = tid & 31;
    const int wv   = tid >> 5;

    const int y0 = yt * 16;
    const int l0 = lc * 512 + wv * 64;
    const int row   = lane & 15;
    const int khalf = (lane >> 4) * 16;

    v8f c0 = {}, c1 = {}, c2 = {}, c3 = {};
    const half_t* aBase = U_half + (size_t)(y0 + row) * FF;
#pragma unroll
    for (int kk = 0; kk < 8; ++kk) {
        const int k = kk * 32;
        v16h a = *(const v16h*)(aBase + k + khalf);
#pragma unroll
        for (int tt = 0; tt < 4; ++tt) {
            const int lcol = l0 + tt * 16 + row;
            v16h bfrag = *(const v16h*)&h_half[((size_t)b * LL + lcol) * FF + k + khalf];
            v8f cc = (tt == 0) ? c0 : (tt == 1) ? c1 : (tt == 2) ? c2 : c3;
            cc = wmma_f16(a, bfrag, cc);
            if (tt == 0) c0 = cc; else if (tt == 1) c1 = cc;
            else if (tt == 2) c2 = cc; else c3 = cc;
        }
    }
#pragma unroll
    for (int tt = 0; tt < 4; ++tt) {
        v8f cc = (tt == 0) ? c0 : (tt == 1) ? c1 : (tt == 2) ? c2 : c3;
        const int l = l0 + tt * 16 + row;
#pragma unroll
        for (int j = 0; j < 8; ++j) {
            const int y = y0 + j + ((lane >> 4) << 3);
            if (y < YY)
                alpha[((size_t)b * YY + y) * LL + l] = cc[j];
        }
    }
}

// ---------------------------------------------------------------------------
// K3: softmax over L (in place on alpha). one block per (b,y) row.
// ---------------------------------------------------------------------------
__global__ void cap_softmax(float* __restrict__ alpha) {
    float* p = alpha + (size_t)blockIdx.x * LL;
    const int tid = threadIdx.x;
    __shared__ float red[256];

    float v[8];
    float mx = -3.0e38f;
#pragma unroll
    for (int i = 0; i < 8; ++i) { v[i] = p[tid + i * 256]; mx = fmaxf(mx, v[i]); }
    red[tid] = mx; __syncthreads();
    for (int s = 128; s > 0; s >>= 1) {
        if (tid < s) red[tid] = fmaxf(red[tid], red[tid + s]);
        __syncthreads();
    }
    mx = red[0]; __syncthreads();

    float sum = 0.0f;
#pragma unroll
    for (int i = 0; i < 8; ++i) { v[i] = __expf(v[i] - mx); sum += v[i]; }
    red[tid] = sum; __syncthreads();
    for (int s = 128; s > 0; s >>= 1) {
        if (tid < s) red[tid] += red[tid + s];
        __syncthreads();
    }
    const float inv = 1.0f / red[0];
#pragma unroll
    for (int i = 0; i < 8; ++i) p[tid + i * 256] = v[i] * inv;
}

// ---------------------------------------------------------------------------
// K4: m[b,y,f] = sum_l alpha[b,y,l] * h[b,l,f]   (M=16 tile, N=FF, K=LL)
// fused head: yhat[b,y] = sum_f final_w[y,f]*m + final_b[y]
// block 256 thr (8 waves), one (b, y-tile) per block; wave owns 32 f cols.
// ---------------------------------------------------------------------------
__global__ void cap_pool(const float* __restrict__ alpha,
                         const half_t* __restrict__ hT_half,
                         const float* __restrict__ final_w,
                         const float* __restrict__ final_b,
                         float* __restrict__ yhat) {
    __shared__ float m_s[16][FF];
    __shared__ float part_s[16][16];

    const int bid = blockIdx.x;
    const int yt  = bid % YTILES;
    const int b   = bid / YTILES;
    const int y0  = yt * 16;
    const int tid  = threadIdx.x;
    const int lane = tid & 31;
    const int wv   = tid >> 5;

    const int row   = lane & 15;
    const int khalf = (lane >> 4) * 16;
    const int f0    = wv * 32;

    const int y  = y0 + row;
    const int yc = (y < YY) ? y : (YY - 1);      // pad rows clamp; stores guarded
    const float* aRow = alpha + ((size_t)b * YY + yc) * LL;

    v8f c0 = {}, c1 = {};
    for (int ks = 0; ks < 64; ++ks) {
        const int l = ks * 32;
        v16h a;
        const float* ap = aRow + l + khalf;       // only 4B-aligned base: scalar loads
#pragma unroll
        for (int i = 0; i < 16; ++i) a[i] = (half_t)ap[i];
#pragma unroll
        for (int tt = 0; tt < 2; ++tt) {
            const int fc = f0 + tt * 16 + row;
            v16h bfrag = *(const v16h*)&hT_half[((size_t)b * FF + fc) * LL + l + khalf];
            if (tt == 0) c0 = wmma_f16(a, bfrag, c0);
            else         c1 = wmma_f16(a, bfrag, c1);
        }
    }
#pragma unroll
    for (int tt = 0; tt < 2; ++tt) {
        v8f cc = (tt == 0) ? c0 : c1;
        const int f = f0 + tt * 16 + row;
#pragma unroll
        for (int j = 0; j < 8; ++j)
            m_s[j + ((lane >> 4) << 3)][f] = cc[j];
    }
    __syncthreads();

    // head: thread t -> (row = t>>4, 16-wide f segment = t&15)
    {
        const int r   = tid >> 4;
        const int seg = tid & 15;
        const int yy  = y0 + r;
        float partial = 0.0f;
        if (yy < YY) {
            const float* fw = final_w + (size_t)yy * FF + seg * 16;
#pragma unroll
            for (int i = 0; i < 16; ++i) partial += fw[i] * m_s[r][seg * 16 + i];
        }
        part_s[r][seg] = partial;
    }
    __syncthreads();
    if (tid < 16) {
        const int yy = y0 + tid;
        if (yy < YY) {
            float s = 0.0f;
#pragma unroll
            for (int i = 0; i < 16; ++i) s += part_s[tid][i];
            yhat[(size_t)b * YY + yy] = s + final_b[yy];
        }
    }
}

// ---------------------------------------------------------------------------
// K5: BCE-with-logits mean loss, single block.
// ---------------------------------------------------------------------------
__global__ void cap_loss(const float* __restrict__ yhat,
                         const float* __restrict__ target,
                         float* __restrict__ loss_out) {
    __shared__ float red[256];
    const int tid = threadIdx.x;
    float acc = 0.0f;
    for (int i = tid; i < NYH; i += 256) {
        float yv = yhat[i];
        float tv = target[i];
        acc += fmaxf(yv, 0.0f) - yv * tv + log1pf(__expf(-fabsf(yv)));
    }
    red[tid] = acc; __syncthreads();
    for (int s = 128; s > 0; s >>= 1) {
        if (tid < s) red[tid] += red[tid + s];
        __syncthreads();
    }
    if (tid == 0) loss_out[0] = red[0] / (float)NYH;
}

// ---------------------------------------------------------------------------
extern "C" void kernel_launch(void* const* d_in, const int* in_sizes, int n_in,
                              void* d_out, int out_size, void* d_ws, size_t ws_size,
                              hipStream_t stream) {
    const int*   x        = (const int*)  d_in[0];
    const float* target   = (const float*)d_in[1];
    const float* embed_W  = (const float*)d_in[2];
    const float* conv_w   = (const float*)d_in[3];
    const float* conv_b   = (const float*)d_in[4];
    const float* U_w      = (const float*)d_in[5];
    const float* final_w  = (const float*)d_in[6];
    const float* final_b  = (const float*)d_in[7];

    float* out   = (float*)d_out;
    float* yhat  = out;                 // [B*Y]
    float* loss  = out + NYH;           // [1]
    float* alpha = out + ALPHA_OFF;     // [B*Y*L]

    half_t* U_half  = (half_t*)d_ws;                       // YPAD*FF
    half_t* wBT     = U_half + (size_t)YPAD * FF;          // FF*KPAD
    half_t* h_half  = wBT + (size_t)FF * KPAD;             // B*L*F
    half_t* hT_half = h_half + (size_t)BB * LL * FF;       // B*F*L

    {   // prep
        int total = YPAD * FF + FF * KPAD;
        cap_prep<<<(total + 255) / 256, 256, 0, stream>>>(U_w, conv_w, U_half, wBT);
    }
    cap_conv<<<BB * (LL / 32), 256, 0, stream>>>(x, embed_W, conv_b, wBT, h_half, hT_half);
    cap_scores<<<BB * YTILES * 4, 256, 0, stream>>>(U_half, h_half, alpha);
    cap_softmax<<<BB * YY, 256, 0, stream>>>(alpha);
    cap_pool<<<BB * YTILES, 256, 0, stream>>>(alpha, hT_half, final_w, final_b, yhat);
    cap_loss<<<1, 256, 0, stream>>>(yhat, target, loss);
}
